// SimpleLTCCell_83262236000491
// MI455X (gfx1250) — matrix operand order repaired and checked by reading, not compiled, not run on hardware
//
#include <hip/hip_runtime.h>
#include <hip/hip_bf16.h>
#include <math.h>

typedef __bf16 bf16_t;
typedef __attribute__((ext_vector_type(16))) __bf16 bf16x16;
typedef __attribute__((ext_vector_type(8)))  __bf16 bf16x8;
typedef __attribute__((ext_vector_type(4)))  __bf16 bf16x4;
typedef __attribute__((ext_vector_type(8)))  float  f32x8;
typedef __attribute__((ext_vector_type(4)))  float  f32x4;
typedef __attribute__((ext_vector_type(4)))  int    i32x4;

#define AS_GLOBAL __attribute__((address_space(1)))
#define AS_LDS    __attribute__((address_space(3)))

// Problem sizes (fixed by the reference): B=128, T=512, D=128, H=512
#define LTC_B 128
#define LTC_T 512
#define LTC_D 128
#define LTC_H 512
#define SCAN_BLOCKS 32

// CDNA5 async global->LDS path (ASYNCcnt), with guarded fallback.
#if defined(__has_builtin)
#if __has_builtin(__builtin_amdgcn_global_load_async_to_lds_b128)
#define LTC_HAVE_ASYNC_LDS 1
#endif
#endif

// ---------------------------------------------------------------------------
// Init: seed bf16 copy of h0 and reset the grid-barrier state (workspace is
// poisoned between phases, so barrier counters MUST be re-initialized here).
// ---------------------------------------------------------------------------
__global__ void ltc_init(const float* __restrict__ hin, bf16_t* __restrict__ hbf,
                         unsigned* __restrict__ bar) {
  const int i = blockIdx.x * blockDim.x + threadIdx.x;
  const int total = LTC_B * LTC_H;
  for (int k = i; k < total; k += gridDim.x * blockDim.x)
    hbf[k] = (bf16_t)hin[k];
  if (i == 0) { bar[0] = 0u; bar[1] = 0u; }
}

// ---------------------------------------------------------------------------
// Kernel 1: x_in[B*T, H] = x[B*T, D] @ W_in^T + b   (written into d_out h_seq)
// Block tile 64(M) x 64(N), full K=128 staged in LDS as bf16, 8 waves,
// each wave: one 16x16 M-tile across two N-tiles, 4 WMMA k-steps.
// ---------------------------------------------------------------------------
__global__ __launch_bounds__(256)
void ltc_inproj(const float* __restrict__ x, const float* __restrict__ Win,
                const float* __restrict__ bin, float* __restrict__ xout) {
  constexpr int K = LTC_D;   // 128
  constexpr int N = LTC_H;   // 512
  __shared__ bf16_t As[64 * K];
  __shared__ bf16_t Bs[64 * K];

  const int tid = threadIdx.x;
  const long m0 = (long)(blockIdx.x >> 3) * 64;
  const int  n0 = (blockIdx.x & 7) * 64;

  // Stage A (x) and B (W_in) tiles, fp32 -> bf16. 2048 float4 each, 8/thread.
#pragma unroll
  for (int i = 0; i < 8; ++i) {
    const int f = tid + 256 * i;
    const int row = f >> 5;
    const int c4 = (f & 31) * 4;
    f32x4 va = *(const f32x4*)(x + (m0 + row) * K + c4);
    f32x4 vb = *(const f32x4*)(Win + (long)(n0 + row) * K + c4);
    bf16x4 pa, pb;
#pragma unroll
    for (int j = 0; j < 4; ++j) { pa[j] = (bf16_t)va[j]; pb[j] = (bf16_t)vb[j]; }
    *(bf16x4*)&As[row * K + c4] = pa;
    *(bf16x4*)&Bs[row * K + c4] = pb;
  }
  __syncthreads();

  const int lane = tid & 31, wv = tid >> 5;
  const int mi = wv & 3;            // M sub-tile (0..3)
  const int np = (wv >> 2) * 2;     // first of two N sub-tiles
  const int lr = lane & 15, hi = lane >> 4;

  f32x8 acc0 = {}; f32x8 acc1 = {};
#pragma unroll
  for (int ks = 0; ks < 4; ++ks) {
    const int kb = ks * 32;
    // A fragment (16x32): lane lr row, K halves {kb+hi*8 .. +7} and {+16 ..}
    const bf16_t* ap = &As[(mi * 16 + lr) * K + kb + hi * 8];
    bf16x8 alo = *(const bf16x8*)ap;
    bf16x8 ahi = *(const bf16x8*)(ap + 16);
    bf16x16 af;
#pragma unroll
    for (int j = 0; j < 8; ++j) { af[j] = alo[j]; af[j + 8] = ahi[j]; }
#pragma unroll
    for (int j = 0; j < 2; ++j) {
      // B fragment (32x16): lane lr col, K halves {kb+hi*8 .. +7} and {+16 ..}
      const bf16_t* bp = &Bs[((np + j) * 16 + lr) * K + kb + hi * 8];
      bf16x8 blo = *(const bf16x8*)bp;
      bf16x8 bhi = *(const bf16x8*)(bp + 16);
      bf16x16 bfv;
#pragma unroll
      for (int q = 0; q < 8; ++q) { bfv[q] = blo[q]; bfv[q + 8] = bhi[q]; }
      if (j == 0)
        acc0 = __builtin_amdgcn_wmma_f32_16x16x32_bf16(false, af, false, bfv,
                                                       (short)0, acc0, false, false);
      else
        acc1 = __builtin_amdgcn_wmma_f32_16x16x32_bf16(false, af, false, bfv,
                                                       (short)0, acc1, false, false);
    }
  }

  // Epilogue: bias (uniform per lane: N = lane&15) + store fp32 x_in.
#pragma unroll
  for (int j = 0; j < 2; ++j) {
    const int n = n0 + (np + j) * 16 + lr;
    const float bias = bin[n];
    f32x8 acc = (j == 0) ? acc0 : acc1;
#pragma unroll
    for (int v = 0; v < 8; ++v) {
      const long m = m0 + mi * 16 + v + 8 * hi;
      xout[m * N + n] = acc[v] + bias;
    }
  }
}

// ---------------------------------------------------------------------------
// Kernel 2: persistent recurrent scan. 32 blocks, block = 32(batch) x 64(col)
// tile of h. W_rec slice (64x512 bf16) resident in LDS for all 512 steps.
// Per step: stage h slice (bf16) -> LDS (async if available), 16 WMMA
// k-steps/wave, tanh + Euler in fp32 registers, write h_seq (fp32, over x_in)
// + h bf16, then grid barrier. Next-step x_in is prefetched in the epilogue.
// ---------------------------------------------------------------------------
__global__ __launch_bounds__(256)
void ltc_scan(const float* __restrict__ hin, const float* __restrict__ tau,
              const float* __restrict__ Wrec, float* __restrict__ out,
              bf16_t* __restrict__ hbf, unsigned* __restrict__ bar) {
  constexpr int H = LTC_H, T = LTC_T;
  extern __shared__ bf16_t smem[];
  bf16_t* Ws = smem;             // [64][512] bf16 = 64 KB (persistent W slice)
  bf16_t* Hs = smem + 64 * H;    // [32][512] bf16 = 32 KB (per-step h slice)

  const int tid = threadIdx.x, lane = tid & 31, wv = tid >> 5;
  const int b0 = (blockIdx.x >> 3) * 32;   // batch-row base
  const int n0 = (blockIdx.x & 7) * 64;    // output-col base
  const int mi = wv & 1, ni = wv >> 1;     // 2 x 4 wave tile grid
  const int lr = lane & 15, hi = lane >> 4;
  const int n = n0 + ni * 16 + lr;

  // Load W_rec slice [n0..n0+63][0..511] fp32 -> bf16 LDS, once.
#pragma unroll
  for (int i = 0; i < 32; ++i) {
    const int f = tid + 256 * i;        // 8192 float4
    const int row = f >> 7;
    const int c4 = (f & 127) * 4;
    f32x4 v = *(const f32x4*)(Wrec + (long)(n0 + row) * H + c4);
    bf16x4 p;
#pragma unroll
    for (int j = 0; j < 4; ++j) p[j] = (bf16_t)v[j];
    *(bf16x4*)&Ws[row * H + c4] = p;
  }

  // Persistent per-lane fp32 state: 8 h values + 1/tau per owned row.
  float hreg[8], itau[8];
  int brow[8];
#pragma unroll
  for (int v = 0; v < 8; ++v) {
    const int b = b0 + mi * 16 + v + 8 * hi;
    brow[v] = b;
    hreg[v] = hin[b * H + n];
    itau[v] = 1.0f / tau[b];
  }
  __syncthreads();

  unsigned gen_seen =
      __hip_atomic_load(&bar[1], __ATOMIC_RELAXED, __HIP_MEMORY_SCOPE_AGENT);

  for (int t = 0; t < T; ++t) {
    // Stage h slice (rows b0..b0+31, bf16) into LDS: 2048 x 16B, 8/thread.
#if defined(LTC_HAVE_ASYNC_LDS)
#pragma unroll
    for (int i = 0; i < 8; ++i) {
      const int c = tid + 256 * i;
      const int row = c >> 6;
      const int c8 = (c & 63) * 8;
      __builtin_amdgcn_global_load_async_to_lds_b128(
          (AS_GLOBAL i32x4*)(hbf + (b0 + row) * H + c8),
          (AS_LDS i32x4*)&Hs[row * H + c8],
          /*offset=*/0, /*cpol=*/0);
    }
    asm volatile("s_wait_asynccnt 0x0" ::: "memory");
#else
#pragma unroll
    for (int i = 0; i < 8; ++i) {
      const int c = tid + 256 * i;
      const int row = c >> 6;
      const int c8 = (c & 63) * 8;
      *(bf16x8*)&Hs[row * H + c8] = *(const bf16x8*)(hbf + (b0 + row) * H + c8);
    }
#endif
    __syncthreads();

    f32x8 acc = {};
#pragma unroll 4
    for (int kk = 0; kk < 16; ++kk) {
      const int kb = kk * 32;
      const bf16_t* ap = &Hs[(mi * 16 + lr) * H + kb + hi * 8];
      bf16x8 alo = *(const bf16x8*)ap;
      bf16x8 ahi = *(const bf16x8*)(ap + 16);
      bf16x16 af;
#pragma unroll
      for (int j = 0; j < 8; ++j) { af[j] = alo[j]; af[j + 8] = ahi[j]; }
      const bf16_t* bp = &Ws[(ni * 16 + lr) * H + kb + hi * 8];
      bf16x8 blo = *(const bf16x8*)bp;
      bf16x8 bhi = *(const bf16x8*)(bp + 16);
      bf16x16 bfv;
#pragma unroll
      for (int j = 0; j < 8; ++j) { bfv[j] = blo[j]; bfv[j + 8] = bhi[j]; }
      acc = __builtin_amdgcn_wmma_f32_16x16x32_bf16(false, af, false, bfv,
                                                    (short)0, acc, false, false);
    }

    // Epilogue: upd = tanh(x_in + h@W^T); h' = h + (upd - h)/tau.
    // x_in lives at the h_seq slot we overwrite (read-then-write, same thread).
#pragma unroll
    for (int v = 0; v < 8; ++v) {
      const int b = brow[v];
      const size_t idx = ((size_t)b * T + t) * H + n;
      const float xin = out[idx];
      const float upd = tanhf(xin + acc[v]);
      const float hn = hreg[v] + (upd - hreg[v]) * itau[v];
      hreg[v] = hn;
      out[idx] = hn;
      hbf[b * H + n] = (bf16_t)hn;
      // Pull next timestep's x_in line toward the WGP (idx(t+1) = idx(t)+H);
      // at t=T-1 this touches the in-bounds h_last region (harmless).
      __builtin_prefetch(&out[idx + H], 0, 3);
    }

    // Grid-wide barrier (32 blocks, all resident): device-scope atomics.
    __threadfence();
    __syncthreads();
    if (tid == 0) {
      const unsigned old = __hip_atomic_fetch_add(&bar[0], 1u, __ATOMIC_ACQ_REL,
                                                  __HIP_MEMORY_SCOPE_AGENT);
      if (old == SCAN_BLOCKS - 1) {
        __hip_atomic_store(&bar[0], 0u, __ATOMIC_RELAXED, __HIP_MEMORY_SCOPE_AGENT);
        __hip_atomic_fetch_add(&bar[1], 1u, __ATOMIC_RELEASE, __HIP_MEMORY_SCOPE_AGENT);
      } else {
        while (__hip_atomic_load(&bar[1], __ATOMIC_ACQUIRE,
                                 __HIP_MEMORY_SCOPE_AGENT) == gen_seen)
          __builtin_amdgcn_s_sleep(1);
      }
      gen_seen++;
    }
    __syncthreads();
  }

  // h_last (fp32) appended after h_seq in d_out.
  const size_t lastoff = (size_t)LTC_B * T * H;
#pragma unroll
  for (int v = 0; v < 8; ++v)
    out[lastoff + (size_t)brow[v] * H + n] = hreg[v];
}

// ---------------------------------------------------------------------------
extern "C" void kernel_launch(void* const* d_in, const int* in_sizes, int n_in,
                              void* d_out, int out_size, void* d_ws, size_t ws_size,
                              hipStream_t stream) {
  (void)in_sizes; (void)n_in; (void)out_size; (void)ws_size;
  const float* x    = (const float*)d_in[0];   // [B,T,D]
  const float* h0   = (const float*)d_in[1];   // [B,H]
  const float* tau  = (const float*)d_in[2];   // [B,1]
  const float* Winw = (const float*)d_in[3];   // [H,D]
  const float* Winb = (const float*)d_in[4];   // [H]
  const float* Wrec = (const float*)d_in[5];   // [H,H]
  float* out = (float*)d_out;                  // h_seq [B,T,H] ++ h_last [B,H]

  unsigned* bar = (unsigned*)d_ws;                      // 2 counters
  bf16_t*   hbf = (bf16_t*)((char*)d_ws + 256);         // bf16 h state [B,H]

  ltc_init<<<64, 256, 0, stream>>>(h0, hbf, bar);

  // Input projection: 65536/64 M-blocks x 512/64 N-blocks = 8192 blocks.
  ltc_inproj<<<dim3(1024 * 8), 256, 0, stream>>>(x, Winw, Winb, out);

  // Persistent scan: 32 blocks, 98304 B dynamic LDS (64KB W + 32KB h).
  ltc_scan<<<dim3(SCAN_BLOCKS), 256, (64 * LTC_H + 32 * LTC_H) * sizeof(bf16_t),
             stream>>>(h0, tau, Wrec, out, hbf, bar);
}
